// NKJ_AI_GPT_71502615544358
// MI455X (gfx1250) — matrix operand order repaired and checked by reading, not compiled, not run on hardware
//
#include <hip/hip_runtime.h>
#include <hip/hip_bf16.h>
#include <math.h>
#include <stdint.h>

// ---------------------------------------------------------------------------
// GPT-2-ish forward (L=12, D=768, H=12, HD=64, F=3072, S=512, B=4, V=50000)
// GEMMs: bf16 WMMA (v_wmma_f32_16x16x32_bf16) with fp32 accumulate.
// Tile staging: Tensor Data Mover (tensor_load_to_lds) double-buffered,
// synchronized with s_wait_tensorcnt + workgroup barrier.
// ---------------------------------------------------------------------------

typedef __bf16 v16bf __attribute__((ext_vector_type(16)));
typedef __bf16 v8bf  __attribute__((ext_vector_type(8)));
typedef float  v8f   __attribute__((ext_vector_type(8)));
typedef unsigned int v4ui __attribute__((ext_vector_type(4)));
typedef int          v8si __attribute__((ext_vector_type(8)));
typedef int          v4si __attribute__((ext_vector_type(4)));

union FragBF16 { v16bf v; v8bf h[2]; };

__device__ __forceinline__ unsigned short f2bf(float f) {
    unsigned int u = __float_as_uint(f);
    unsigned int r = u + 0x7FFFu + ((u >> 16) & 1u);   // round-to-nearest-even
    return (unsigned short)(r >> 16);
}

// ---------------------------------------------------------------------------
// TDM: issue a 2-D tile load (bf16 elements) from global to LDS.
// D# per CDNA5 ISA ch.8: group0 = {count, lds_addr, global_addr, type=2},
// group1 = {data_size=2B, tensor_dim0/1, tile_dim0/1, tensor_dim0_stride}.
// ---------------------------------------------------------------------------
__device__ __forceinline__ void tdm_load_tile_2d(
    unsigned lds_off, const void* gptr,
    unsigned td0, unsigned td1, unsigned stride0,
    unsigned tile0, unsigned tile1)
{
    unsigned long long ga = (unsigned long long)(uintptr_t)gptr;
    v4ui g0;
    g0.x = 1u;                                            // count=1 (valid D#)
    g0.y = lds_off;                                       // lds_addr [63:32]
    g0.z = (unsigned)(ga & 0xFFFFFFFFu);                  // global_addr[31:0]
    g0.w = (unsigned)((ga >> 32) & 0x1FFFFFFu) | (2u << 30); // addr[56:32]|type=2
    v8si g1;
    g1[0] = (int)(1u << 16);                              // data_size=1 -> 2 bytes
    g1[1] = (int)((td0 & 0xFFFFu) << 16);                 // tensor_dim0[15:0]
    g1[2] = (int)(((td0 >> 16) & 0xFFFFu) | ((td1 & 0xFFFFu) << 16));
    g1[3] = (int)(((td1 >> 16) & 0xFFFFu) | ((tile0 & 0xFFFFu) << 16));
    g1[4] = (int)(tile1 & 0xFFFFu);                       // tile_dim1, tile_dim2=0
    g1[5] = (int)stride0;                                 // tensor_dim0_stride[31:0]
    g1[6] = 0;
    g1[7] = 0;
    v4si gz = {0, 0, 0, 0};
#if defined(__clang_major__) && (__clang_major__ >= 23)
    v8si gz8 = {0, 0, 0, 0, 0, 0, 0, 0};
    __builtin_amdgcn_tensor_load_to_lds(g0, g1, gz, gz, gz8, 0);
#else
    __builtin_amdgcn_tensor_load_to_lds(g0, g1, gz, gz, 0);
#endif
}

// ---------------------------------------------------------------------------
// WMMA GEMM:  C[M,N] = epilogue(A[M,K] @ B[K,N] + bias [+ res])
//   A  : bf16 [M,K] row-major (produced bf16 by LN / GELU / attention)
//   Bt : bf16 [N,K] row-major (= B^T, pre-transposed+converted weights)
// Block tile 128x128, BK=64, 256 thr = 8 waves (2M x 4N), 4x2 accs/wave.
// TDM double-buffered staging; fragments via contiguous ds_load_b128.
// EPI: 0 = +bias -> fp32 ; 1 = +bias+res -> fp32 ; 2 = gelu(+bias) -> bf16
// Requires M%128==0, K%64==0; N ragged OK (store-guarded).
// ---------------------------------------------------------------------------
#define BM 128
#define BN 128
#define BK 64

template <int EPI>
__global__ __launch_bounds__(256) void gemm_bf16_wmma(
    const unsigned short* __restrict__ A,
    const unsigned short* __restrict__ Bt,
    const float* __restrict__ bias, const float* __restrict__ res,
    void* __restrict__ Cout, int M, int N, int K)
{
    __shared__ __attribute__((aligned(128))) unsigned short As[2][BM][BK];
    __shared__ __attribute__((aligned(128))) unsigned short Bs[2][BN][BK];

    const int tid  = threadIdx.x;
    const int lane = tid & 31;
    const int wave = tid >> 5;
    const int wm   = wave & 1;    // 64 rows per wave-row
    const int wn   = wave >> 1;   // 32 cols per wave-col
    const int blockM = blockIdx.y * BM;
    const int blockN = blockIdx.x * BN;

    const unsigned ldsA = (unsigned)(uintptr_t)&As[0][0][0];
    const unsigned ldsB = (unsigned)(uintptr_t)&Bs[0][0][0];
    const unsigned bufA = BM * BK * 2;   // bytes per A buffer
    const unsigned bufB = BN * BK * 2;

    v8f acc[4][2];
    {
        v8f z = {0.f, 0.f, 0.f, 0.f, 0.f, 0.f, 0.f, 0.f};
#pragma unroll
        for (int i = 0; i < 4; ++i)
#pragma unroll
            for (int j = 0; j < 2; ++j) acc[i][j] = z;
    }

    // ---- prologue: DMA first K-slab into buffer 0 ----
    if (wave == 0) {
        tdm_load_tile_2d(ldsA, A + (size_t)blockM * K,
                         (unsigned)K, (unsigned)(M - blockM), (unsigned)K, BK, BM);
        tdm_load_tile_2d(ldsB, Bt + (size_t)blockN * K,
                         (unsigned)K, (unsigned)(N - blockN), (unsigned)K, BK, BN);
    }
    __builtin_amdgcn_s_wait_tensorcnt(0);
    __syncthreads();

    const int arow = wm * 64 + (lane & 15);
    const int akb  = (lane >> 4) * 16;           // A frag K chunk base
    const int bcol = wn * 32 + (lane & 15);
    const int bk0  = (lane >> 4) * 16;           // B frag K base

    int buf = 0;
    for (int kb = 0; kb < K; kb += BK) {
        int nxt = kb + BK;
        if (nxt < K && wave == 0) {              // DMA next slab into other buffer
            tdm_load_tile_2d(ldsA + (buf ^ 1) * bufA, A + (size_t)blockM * K + nxt,
                             (unsigned)(K - nxt), (unsigned)(M - blockM),
                             (unsigned)K, BK, BM);
            tdm_load_tile_2d(ldsB + (buf ^ 1) * bufB, Bt + (size_t)blockN * K + nxt,
                             (unsigned)(K - nxt), (unsigned)(N - blockN),
                             (unsigned)K, BK, BN);
        }
        // ---- compute on current buffer: 2 k-steps x 8 WMMAs ----
#pragma unroll
        for (int ks = 0; ks < BK; ks += 32) {
            FragBF16 af[4], bf[2];
#pragma unroll
            for (int i = 0; i < 4; ++i) {
                const unsigned short* rp = &As[buf][arow + i * 16][ks];
                af[i].h[0] = *(const v8bf*)(rp + akb);       // K = akb..akb+7
                af[i].h[1] = *(const v8bf*)(rp + akb + 8);   // K = akb+8..akb+15
            }
#pragma unroll
            for (int j = 0; j < 2; ++j) {
                const unsigned short* cp = &Bs[buf][bcol + j * 16][ks];
                bf[j].h[0] = *(const v8bf*)(cp + bk0);       // K = bk0..bk0+7
                bf[j].h[1] = *(const v8bf*)(cp + bk0 + 8);   // K = bk0+8..bk0+15
            }
#pragma unroll
            for (int i = 0; i < 4; ++i)
#pragma unroll
                for (int j = 0; j < 2; ++j)
                    acc[i][j] = __builtin_amdgcn_wmma_f32_16x16x32_bf16(
                        false, af[i].v, false, bf[j].v,
                        (short)0, acc[i][j], false, false);
        }
        __builtin_amdgcn_s_wait_tensorcnt(0);    // next slab landed
        __syncthreads();                         // everyone done reading `buf`
        buf ^= 1;
    }

    // ---- epilogue: C frag -> row = r + 8*(lane>=16), col = lane&15 ----
    const int rbase = blockM + wm * 64;
    const int cbase = blockN + wn * 32;
    const int rhalf = (lane >> 4) * 8;
    float* Cf = (float*)Cout;
    unsigned short* Cb = (unsigned short*)Cout;
#pragma unroll
    for (int i = 0; i < 4; ++i) {
#pragma unroll
        for (int j = 0; j < 2; ++j) {
            int col = cbase + j * 16 + (lane & 15);
            if (col >= N) continue;
            float bb = bias[col];
#pragma unroll
            for (int r = 0; r < 8; ++r) {
                int row = rbase + i * 16 + rhalf + r;
                float val = acc[i][j][r] + bb;
                if (EPI == 1) val += res[(size_t)row * N + col];
                if (EPI == 2) {
                    val = 0.5f * val * (1.0f + erff(val * 0.70710678118654752f));
                    Cb[(size_t)row * N + col] = f2bf(val);
                } else {
                    Cf[(size_t)row * N + col] = val;
                }
            }
        }
    }
}

// ---------------------------------------------------------------------------
// Weight transpose + fp32 -> bf16 convert:  W[K,N] fp32  ->  Bt[N,K] bf16
// ---------------------------------------------------------------------------
__global__ __launch_bounds__(256) void transpose_bf16_kernel(
    const float* __restrict__ W, unsigned short* __restrict__ Bt, int K, int N)
{
    __shared__ float t[32][33];
    int nt = blockIdx.x * 32, kt = blockIdx.y * 32;
    int tx = threadIdx.x & 31, ty = threadIdx.x >> 5;    // ty 0..7
#pragma unroll
    for (int i = 0; i < 4; ++i) {
        int k = kt + ty + i * 8, n = nt + tx;
        t[ty + i * 8][tx] = (k < K && n < N) ? W[(size_t)k * N + n] : 0.f;
    }
    __syncthreads();
#pragma unroll
    for (int i = 0; i < 4; ++i) {
        int n = nt + ty + i * 8, k = kt + tx;
        if (n < N && k < K) Bt[(size_t)n * K + k] = f2bf(t[tx][ty + i * 8]);
    }
}

// ---------------------------------------------------------------------------
// Embedding gather + positional add (fp32 residual stream)
// ---------------------------------------------------------------------------
__global__ __launch_bounds__(256) void embed_kernel(
    const int* __restrict__ ids, const float* __restrict__ tok,
    const float* __restrict__ pos, float* __restrict__ x, int S, int D)
{
    int row = blockIdx.x;
    int s   = row % S;
    int id  = ids[row];
    const float* tp = tok + (size_t)id * D;
    const float* pp = pos + (size_t)s * D;
    float* xp = x + (size_t)row * D;
    for (int i = threadIdx.x; i < D; i += 256) xp[i] = tp[i] + pp[i];
}

// ---------------------------------------------------------------------------
// LayerNorm (fp32 in) -> bf16 out (feeds WMMA GEMM A operand)
// ---------------------------------------------------------------------------
__global__ __launch_bounds__(256) void ln_kernel(
    const float* __restrict__ x, const float* __restrict__ s,
    const float* __restrict__ b, unsigned short* __restrict__ y, int D)
{
    __shared__ float red[256];
    int row = blockIdx.x;
    int tid = threadIdx.x;
    const float* xp = x + (size_t)row * D;

    float sum = 0.f, sumsq = 0.f;
    for (int i = tid; i < D; i += 256) {
        float v = xp[i];
        sum += v; sumsq += v * v;
    }
    red[tid] = sum;  __syncthreads();
    for (int st = 128; st > 0; st >>= 1) {
        if (tid < st) red[tid] += red[tid + st];
        __syncthreads();
    }
    float mean = red[0] / (float)D;
    __syncthreads();
    red[tid] = sumsq; __syncthreads();
    for (int st = 128; st > 0; st >>= 1) {
        if (tid < st) red[tid] += red[tid + st];
        __syncthreads();
    }
    float var = red[0] / (float)D - mean * mean;
    float inv = rsqrtf(var + 1e-5f);

    unsigned short* yp = y + (size_t)row * D;
    for (int i = tid; i < D; i += 256)
        yp[i] = f2bf((xp[i] - mean) * inv * s[i] + b[i]);
}

// ---------------------------------------------------------------------------
// Causal attention, online softmax; one wave per (b,h,q), 2 dims/lane.
// q,k,v fp32 [B,S,D]; ctx out bf16 (A operand of the Wo GEMM).
// ---------------------------------------------------------------------------
__global__ __launch_bounds__(32) void attn_kernel(
    const float* __restrict__ q, const float* __restrict__ k,
    const float* __restrict__ v, unsigned short* __restrict__ ctx,
    int S, int D, int H, int HD)
{
    int bqh = blockIdx.x;          // (b*S + qs)*H + h
    int h   = bqh % H;
    int t   = bqh / H;
    int qs  = t % S;
    int b   = t / S;
    int lane = threadIdx.x;

    size_t rowQ = (size_t)(b * S + qs) * D + h * HD;
    float2 qv = *(const float2*)(q + rowQ + lane * 2);

    float m = -INFINITY, l = 0.f;
    float2 cx = make_float2(0.f, 0.f);

    for (int ks = 0; ks <= qs; ++ks) {
        size_t rowK = (size_t)(b * S + ks) * D + h * HD;
        float2 kv = *(const float2*)(k + rowK + lane * 2);
        float p = qv.x * kv.x + qv.y * kv.y;
#pragma unroll
        for (int off = 16; off > 0; off >>= 1)
            p += __shfl_xor(p, off, 32);
        float sc = p * 0.125f;                        // / sqrt(64)

        float mnew = fmaxf(m, sc);
        float corr = __expf(m - mnew);
        float e    = __expf(sc - mnew);
        l = l * corr + e;
        float2 vv = *(const float2*)(v + rowK + lane * 2);
        cx.x = cx.x * corr + e * vv.x;
        cx.y = cx.y * corr + e * vv.y;
        m = mnew;
    }
    float invl = 1.0f / l;
    unsigned pack = (unsigned)f2bf(cx.x * invl) |
                    ((unsigned)f2bf(cx.y * invl) << 16);
    *(unsigned*)(ctx + rowQ + lane * 2) = pack;
}

// ---------------------------------------------------------------------------
// Host-side orchestration
// ---------------------------------------------------------------------------
extern "C" void kernel_launch(void* const* d_in, const int* in_sizes, int n_in,
                              void* d_out, int out_size, void* d_ws, size_t ws_size,
                              hipStream_t stream) {
    (void)in_sizes; (void)n_in; (void)out_size; (void)ws_size;

    const int Bc = 4, Sc = 512, Dc = 768, Lc = 12, Hc = 12, HDc = 64,
              Fc = 3072, Vc = 50000;
    const int M = Bc * Sc;                 // 2048 token rows

    const int*   ids  = (const int*)  d_in[0];
    const float* tok  = (const float*)d_in[1];
    const float* pos  = (const float*)d_in[2];
    const float* Wq   = (const float*)d_in[3];
    const float* bq   = (const float*)d_in[4];
    const float* Wk   = (const float*)d_in[5];
    const float* bk   = (const float*)d_in[6];
    const float* Wv   = (const float*)d_in[7];
    const float* bv   = (const float*)d_in[8];
    const float* Wo   = (const float*)d_in[9];
    const float* bo   = (const float*)d_in[10];
    const float* ln1s = (const float*)d_in[11];
    const float* ln1b = (const float*)d_in[12];
    const float* W1   = (const float*)d_in[13];
    const float* b1   = (const float*)d_in[14];
    const float* W2   = (const float*)d_in[15];
    const float* b2   = (const float*)d_in[16];
    const float* ln2s = (const float*)d_in[17];
    const float* ln2b = (const float*)d_in[18];
    const float* lnfs = (const float*)d_in[19];
    const float* lnfb = (const float*)d_in[20];
    const float* Wout = (const float*)d_in[21];
    const float* bout = (const float*)d_in[22];
    float* logits = (float*)d_out;

    // workspace layout (byte offsets; all chunks 256B-aligned by construction)
    char* w = (char*)d_ws;
    float* x = (float*)w;                    w += (size_t)M * Dc * 4;   // residual
    unsigned short* hb = (unsigned short*)w; w += (size_t)M * Dc * 2;   // LN out bf16
    float* qb = (float*)w;                   w += (size_t)M * Dc * 4;
    float* kb2 = (float*)w;                  w += (size_t)M * Dc * 4;
    float* vb = (float*)w;                   w += (size_t)M * Dc * 4;
    unsigned short* cb = (unsigned short*)w; w += (size_t)M * Dc * 2;   // ctx bf16
    unsigned short* fb = (unsigned short*)w; w += (size_t)M * Fc * 2;   // gelu bf16
    unsigned short* Bt = (unsigned short*)w;                            // B^T bf16 staging

    dim3 gD(Dc / BN, M / BM);
    dim3 gF(Fc / BN, M / BM);
    dim3 gV((Vc + BN - 1) / BN, M / BM);

    auto xpose = [&](const float* Wp, int K, int N) {
        dim3 g((N + 31) / 32, (K + 31) / 32);
        transpose_bf16_kernel<<<g, 256, 0, stream>>>(Wp, Bt, K, N);
    };

    embed_kernel<<<M, 256, 0, stream>>>(ids, tok, pos, x, Sc, Dc);

    for (int l = 0; l < Lc; ++l) {
        const float* wq = Wq + (size_t)l * Dc * Dc;
        const float* wk = Wk + (size_t)l * Dc * Dc;
        const float* wv = Wv + (size_t)l * Dc * Dc;
        const float* wo = Wo + (size_t)l * Dc * Dc;
        const float* w1 = W1 + (size_t)l * Dc * Fc;
        const float* w2 = W2 + (size_t)l * Fc * Dc;

        ln_kernel<<<M, 256, 0, stream>>>(x, ln1s + l * Dc, ln1b + l * Dc, hb, Dc);

        xpose(wq, Dc, Dc);
        gemm_bf16_wmma<0><<<gD, 256, 0, stream>>>(hb, Bt, bq + l * Dc, nullptr, qb, M, Dc, Dc);
        xpose(wk, Dc, Dc);
        gemm_bf16_wmma<0><<<gD, 256, 0, stream>>>(hb, Bt, bk + l * Dc, nullptr, kb2, M, Dc, Dc);
        xpose(wv, Dc, Dc);
        gemm_bf16_wmma<0><<<gD, 256, 0, stream>>>(hb, Bt, bv + l * Dc, nullptr, vb, M, Dc, Dc);

        attn_kernel<<<M * Hc, 32, 0, stream>>>(qb, kb2, vb, cb, Sc, Dc, Hc, HDc);

        // x = x + ctx @ Wo + bo   (in-place residual)
        xpose(wo, Dc, Dc);
        gemm_bf16_wmma<1><<<gD, 256, 0, stream>>>(cb, Bt, bo + l * Dc, x, x, M, Dc, Dc);

        ln_kernel<<<M, 256, 0, stream>>>(x, ln2s + l * Dc, ln2b + l * Dc, hb, Dc);

        // fb = gelu(h2 @ W1 + b1)  (bf16 out)
        xpose(w1, Dc, Fc);
        gemm_bf16_wmma<2><<<gF, 256, 0, stream>>>(hb, Bt, b1 + l * Fc, nullptr, fb, M, Fc, Dc);
        // x = x + fb @ W2 + b2
        xpose(w2, Fc, Dc);
        gemm_bf16_wmma<1><<<gD, 256, 0, stream>>>(fb, Bt, b2 + l * Dc, x, x, M, Dc, Fc);
    }

    ln_kernel<<<M, 256, 0, stream>>>(x, lnfs, lnfb, hb, Dc);
    xpose(Wout, Dc, Vc);
    gemm_bf16_wmma<0><<<gV, 256, 0, stream>>>(hb, Bt, bout, nullptr, logits, M, Vc, Dc);
}
